// Cross_attention_8942121910949
// MI455X (gfx1250) — matrix operand order repaired and checked by reading, not compiled, not run on hardware
//
#include <hip/hip_runtime.h>

// Problem constants (match reference).
constexpr int kB  = 16;
constexpr int kLD = 1024;
constexpr int kLQ = 512;
constexpr int kH  = 1024;

typedef __attribute__((ext_vector_type(16))) __bf16 v16bf;
typedef __attribute__((ext_vector_type(8)))  float  v8f;

union Frag {
    v16bf v;
    unsigned int u[8];
};

__device__ __forceinline__ unsigned short f2bf(float f) {
    unsigned int u = __float_as_uint(f);
    u += 0x7FFFu + ((u >> 16) & 1u);   // round-to-nearest-even
    return (unsigned short)(u >> 16);
}

// ---------------------------------------------------------------------------
// Tiled bf16 WMMA GEMM:  C[M,N] = A[M,K] * B  (B either [K,N] or [N,K]^T)
// Workgroup: 256 threads = 8 waves, tile 128(M) x 64(N), K chunk = 32.
// Each wave owns a 32x32 subtile = 2x2 v_wmma_f32_16x16x32_bf16 tiles.
// A/B may be f32 (converted to bf16 while staging to LDS) or bf16.
// Output: f32 (strided, with column offset) and optionally bf16 mirror.
// ---------------------------------------------------------------------------
template <bool A_F32, bool B_F32, bool B_T, bool HAS_BF>
__global__ __launch_bounds__(256) void gemm_wmma(
    const void* __restrict__ Ap, const void* __restrict__ Bp,
    float* __restrict__ outF, unsigned short* __restrict__ outBF,
    int K, int lda, int ldb,
    long aBatch, long bBatch,
    long ofBatch, int ldOF, int ofCol,
    long obBatch, int ldOB, int obCol)
{
    constexpr int BM = 128, BN = 64, KC = 32;
    constexpr int ASTR = KC + 2;       // padded LDS row stride (halves)
    constexpr int BSTR = 2 * BN + 2;   // padded paired-K row stride (halves)
    __shared__ unsigned short As[BM * ASTR];
    __shared__ unsigned short Bs[(KC / 2) * BSTR];

    const int t    = threadIdx.x;
    const int b    = blockIdx.z;
    const int m0   = blockIdx.y * BM;
    const int n0   = blockIdx.x * BN;
    const int wave = t >> 5;
    const int lane = t & 31;
    const int wm   = wave >> 1;   // 0..3
    const int wn   = wave & 1;    // 0..1

    const float*          Af = (const float*)Ap;
    const unsigned short* Ah = (const unsigned short*)Ap;
    const float*          Bf = (const float*)Bp;
    const unsigned short* Bh = (const unsigned short*)Bp;

    v8f acc[2][2] = {};

    const int half = lane >> 4;     // 0: lanes 0-15, 1: lanes 16-31
    const int mr   = lane & 15;

    for (int kc = 0; kc < K; kc += KC) {
        // ---- stage A tile: BM x KC ----
#pragma unroll
        for (int i = 0; i < (BM * KC) / 256; i++) {
            int e = i * 256 + t;
            int r = e >> 5, k = e & 31;
            size_t src = (size_t)aBatch * b + (size_t)(m0 + r) * lda + kc + k;
            As[r * ASTR + k] = A_F32 ? f2bf(Af[src]) : Ah[src];
        }
        // ---- stage B tile: KC x BN, paired-K layout ----
#pragma unroll
        for (int i = 0; i < (BN * KC) / 256; i++) {
            int e = i * 256 + t;
            int k, n;
            size_t src;
            if (B_T) { n = e >> 5; k = e & 31;
                       src = (size_t)bBatch * b + (size_t)(n0 + n) * ldb + kc + k; }
            else     { k = e >> 6; n = e & 63;
                       src = (size_t)bBatch * b + (size_t)(kc + k) * ldb + n0 + n; }
            unsigned short h = B_F32 ? f2bf(Bf[src]) : Bh[src];
            Bs[(k >> 1) * BSTR + 2 * n + (k & 1)] = h;
        }
        __syncthreads();

        // ---- build fragments (ISA 16-bit A 16x32 / B 32x16 layouts) ----
        Frag af[2], bfr[2];
#pragma unroll
        for (int sm = 0; sm < 2; sm++) {
            int mb = wm * 32 + sm * 16 + mr;
#pragma unroll
            for (int i = 0; i < 8; i++) {
                int kb = (i < 4 ? 2 * i : 2 * i + 8) + half * 8;
                af[sm].u[i] = *(const unsigned int*)&As[mb * ASTR + kb];
            }
        }
#pragma unroll
        for (int sn = 0; sn < 2; sn++) {
            int nb = wn * 32 + sn * 16 + mr;
#pragma unroll
            for (int i = 0; i < 8; i++) {
                int k = (i < 4 ? 2 * i : 2 * i + 8) + half * 8;
                bfr[sn].u[i] = *(const unsigned int*)&Bs[(k >> 1) * BSTR + 2 * nb];
            }
        }
#pragma unroll
        for (int sm = 0; sm < 2; sm++)
#pragma unroll
            for (int sn = 0; sn < 2; sn++)
                acc[sm][sn] = __builtin_amdgcn_wmma_f32_16x16x32_bf16(
                    false, af[sm].v, false, bfr[sn].v,
                    (short)0, acc[sm][sn], false, false);
        __syncthreads();
    }

    // ---- epilogue: C/D layout row = r + 8*half, col = lane&15 ----
#pragma unroll
    for (int sm = 0; sm < 2; sm++) {
#pragma unroll
        for (int sn = 0; sn < 2; sn++) {
            int gm = m0 + wm * 32 + sm * 16 + half * 8;
            int gn = n0 + wn * 32 + sn * 16 + mr;
#pragma unroll
            for (int r = 0; r < 8; r++) {
                float v = acc[sm][sn][r];
                size_t of = (size_t)ofBatch * b + (size_t)(gm + r) * ldOF + ofCol + gn;
                outF[of] = v;
                if (HAS_BF) {
                    size_t ob = (size_t)obBatch * b + (size_t)(gm + r) * ldOB + obCol + gn;
                    outBF[ob] = f2bf(v);
                }
            }
        }
    }
}

// ---------------------------------------------------------------------------
// Row softmax over q (width LQ=512) with query_mask; outputs bf16 A_d.
// One 256-thread block per (b, d) row; 2 elements per thread.
// ---------------------------------------------------------------------------
__global__ __launch_bounds__(256) void softmax_rows(
    const float* __restrict__ S, const unsigned char* __restrict__ qmask,
    unsigned short* __restrict__ Ad)
{
    const int row = blockIdx.x;          // b*LD + d
    const int b   = row >> 10;           // LD = 1024
    const int t   = threadIdx.x;
    const float* Srow = S + (size_t)row * kLQ;
    __shared__ float red[256];

    const int q0 = t, q1 = t + 256;
    const bool msk0 = qmask[b * kLQ + q0] != 0;
    const bool msk1 = qmask[b * kLQ + q1] != 0;
    float x0 = msk0 ? -3.0e38f : Srow[q0];
    float x1 = msk1 ? -3.0e38f : Srow[q1];

    red[t] = fmaxf(x0, x1);
    __syncthreads();
    for (int s = 128; s > 0; s >>= 1) {
        if (t < s) red[t] = fmaxf(red[t], red[t + s]);
        __syncthreads();
    }
    float mx = red[0];
    __syncthreads();

    float e0 = msk0 ? 0.f : __expf(x0 - mx);
    float e1 = msk1 ? 0.f : __expf(x1 - mx);
    red[t] = e0 + e1;
    __syncthreads();
    for (int s = 128; s > 0; s >>= 1) {
        if (t < s) red[t] += red[t + s];
        __syncthreads();
    }
    float rs = 1.f / red[0];

    unsigned short* Arow = Ad + (size_t)row * kLQ;
    Arow[q0] = f2bf(e0 * rs);
    Arow[q1] = f2bf(e1 * rs);
}

// ---------------------------------------------------------------------------
// Column softmax stats over d (LD) per (b,q) with doc_mask (coalesced reads).
// Block handles 32 q columns; threads laid out 32(q) x 8(d).
// ---------------------------------------------------------------------------
__global__ __launch_bounds__(256) void col_stats(
    const float* __restrict__ S, const unsigned char* __restrict__ dmask,
    float* __restrict__ colmax, float* __restrict__ colrsum)
{
    const int b  = blockIdx.y;
    const int q0 = blockIdx.x * 32;
    const int t  = threadIdx.x;
    const int tq = t & 31, td = t >> 5;
    __shared__ float part[8][32];
    __shared__ float cmax[32];

    const float* Sb = S + (size_t)b * kLD * kLQ;
    const unsigned char* mb = dmask + (size_t)b * kLD;

    float m = -3.0e38f;
    for (int d = td; d < kLD; d += 8)
        if (!mb[d]) m = fmaxf(m, Sb[(size_t)d * kLQ + q0 + tq]);
    part[td][tq] = m;
    __syncthreads();
    if (td == 0) {
        float mm = part[0][tq];
        for (int j = 1; j < 8; j++) mm = fmaxf(mm, part[j][tq]);
        cmax[tq] = mm;
    }
    __syncthreads();
    float mm = cmax[tq];

    float s = 0.f;
    for (int d = td; d < kLD; d += 8)
        if (!mb[d]) s += __expf(Sb[(size_t)d * kLQ + q0 + tq] - mm);
    part[td][tq] = s;
    __syncthreads();
    if (td == 0) {
        float ss = 0.f;
        for (int j = 0; j < 8; j++) ss += part[j][tq];
        colmax[b * kLQ + q0 + tq]  = mm;
        colrsum[b * kLQ + q0 + tq] = 1.f / ss;
    }
}

// ---------------------------------------------------------------------------
// Apply column softmax + transpose via LDS: A_qT[b,q,d] (bf16), coalesced.
// ---------------------------------------------------------------------------
__global__ __launch_bounds__(256) void col_apply(
    const float* __restrict__ S, const unsigned char* __restrict__ dmask,
    const float* __restrict__ colmax, const float* __restrict__ colrsum,
    unsigned short* __restrict__ AqT)
{
    const int b  = blockIdx.z;
    const int d0 = blockIdx.y * 64;
    const int q0 = blockIdx.x * 64;
    const int t  = threadIdx.x;
    __shared__ unsigned short tile[64 * 65];

#pragma unroll
    for (int i = 0; i < 16; i++) {
        int e = i * 256 + t;
        int r = e >> 6, c = e & 63;
        int d = d0 + r, q = q0 + c;
        float v;
        if (dmask[b * kLD + d]) v = 0.f;
        else v = __expf(S[((size_t)b * kLD + d) * kLQ + q] - colmax[b * kLQ + q])
                 * colrsum[b * kLQ + q];
        tile[c * 65 + r] = f2bf(v);
    }
    __syncthreads();
#pragma unroll
    for (int i = 0; i < 16; i++) {
        int e = i * 256 + t;
        int qr = e >> 6, dd = e & 63;
        AqT[((size_t)b * kLQ + q0 + qr) * kLD + d0 + dd] = tile[qr * 65 + dd];
    }
}

// Copy Q into query_output[:, :, 0:H] (f32) and QObf[:, :, 0:H] (bf16).
__global__ __launch_bounds__(256) void copy_q(
    const float* __restrict__ Q, float* __restrict__ qout,
    unsigned short* __restrict__ QObf)
{
    const int row = blockIdx.x;          // b*LQ + q
    const int t = threadIdx.x;
    const float* src = Q + (size_t)row * kH;
    float* dst = qout + (size_t)row * 2 * kH;
    unsigned short* dbf = QObf + (size_t)row * 2 * kH;
    for (int h = t; h < kH; h += 256) {
        float v = src[h];
        dst[h] = v;
        dbf[h] = f2bf(v);
    }
}

// Copy D into doc_output[:, :, 2H:3H].
__global__ __launch_bounds__(256) void copy_d(
    const float* __restrict__ D, float* __restrict__ docout)
{
    const int row = blockIdx.x;          // b*LD + d
    const int t = threadIdx.x;
    const float* src = D + (size_t)row * kH;
    float* dst = docout + (size_t)row * 3 * kH + 2 * kH;
    for (int h = t; h < kH; h += 256) dst[h] = src[h];
}

// Gather last = doc_output[b, doc_lens[b]-1, :].
__global__ __launch_bounds__(256) void gather_last(
    const float* __restrict__ docout, const int* __restrict__ dlens,
    float* __restrict__ last)
{
    const int b = blockIdx.x;
    const int t = threadIdx.x;
    const int r = dlens[b] - 1;
    const float* src = docout + ((size_t)b * kLD + r) * 3 * kH;
    for (int h = t; h < 3 * kH; h += 256) last[b * 3 * kH + h] = src[h];
}

extern "C" void kernel_launch(void* const* d_in, const int* in_sizes, int n_in,
                              void* d_out, int out_size, void* d_ws, size_t ws_size,
                              hipStream_t stream)
{
    (void)in_sizes; (void)n_in; (void)out_size; (void)ws_size;

    const float*         D     = (const float*)d_in[0];
    const float*         Q     = (const float*)d_in[1];
    const unsigned char* dmask = (const unsigned char*)d_in[2];
    const unsigned char* qmask = (const unsigned char*)d_in[3];
    const int*           dlens = (const int*)d_in[4];

    float* out      = (float*)d_out;
    float* doc_out  = out;                                   // [B,LD,3H]
    float* q_out    = out + (size_t)kB * kLD * 3 * kH;       // [B,LQ,2H]
    float* last_out = q_out + (size_t)kB * kLQ * 2 * kH;     // [B,3H]

    char* ws = (char*)d_ws;
    float*          S    = (float*)ws;                        // B*LD*LQ f32 (33.5 MB)
    unsigned short* QObf = (unsigned short*)ws;               // aliases S once S is dead
    unsigned short* Ad   = (unsigned short*)(ws + 33554432);  // B*LD*LQ bf16
    unsigned short* AqT  = (unsigned short*)(ws + 50331648);  // B*LQ*LD bf16
    float*          cmax = (float*)(ws + 67108864);           // B*LQ
    float*          crsm = cmax + kB * kLQ;                   // B*LQ

    dim3 blk(256);

    // 1) S = D @ Q^T   (A=f32, B=f32 transposed)
    gemm_wmma<true, true, true, false><<<dim3(kLQ / 64, kLD / 128, kB), blk, 0, stream>>>(
        D, Q, S, nullptr,
        /*K=*/kH, /*lda=*/kH, /*ldb=*/kH,
        (long)kLD * kH, (long)kLQ * kH,
        (long)kLD * kLQ, kLQ, 0,
        0, 0, 0);

    // 2) A_d = row-softmax(S, query_mask)  -> bf16
    softmax_rows<<<dim3(kB * kLD), blk, 0, stream>>>(S, qmask, Ad);

    // 3) A_qT = col-softmax(S, doc_mask)^T -> bf16
    col_stats<<<dim3(kLQ / 32, kB), blk, 0, stream>>>(S, dmask, cmax, crsm);
    col_apply<<<dim3(kLQ / 64, kLD / 64, kB), blk, 0, stream>>>(S, dmask, cmax, crsm, AqT);

    // 4) query_output[:, 0:H] = Q (f32 + bf16 mirror). S is dead; QObf aliases it.
    copy_q<<<dim3(kB * kLQ), blk, 0, stream>>>(Q, q_out, QObf);

    // 5) C_q = A_qT @ D -> query_output[:, H:2H] + QObf[:, H:2H]
    gemm_wmma<false, true, false, true><<<dim3(kH / 64, kLQ / 128, kB), blk, 0, stream>>>(
        AqT, D, q_out, QObf,
        /*K=*/kLD, /*lda=*/kLD, /*ldb=*/kH,
        (long)kLQ * kLD, (long)kLD * kH,
        (long)kLQ * 2 * kH, 2 * kH, kH,
        (long)kLQ * 2 * kH, 2 * kH, kH);

    // 6) doc_output[:, 2H:3H] = D ; C_D = A_d @ QObf -> doc_output[:, 0:2H]
    copy_d<<<dim3(kB * kLD), blk, 0, stream>>>(D, doc_out);
    gemm_wmma<false, false, false, false><<<dim3(2 * kH / 64, kLD / 128, kB), blk, 0, stream>>>(
        Ad, QObf, doc_out, nullptr,
        /*K=*/kLQ, /*lda=*/kLQ, /*ldb=*/2 * kH,
        (long)kLD * kLQ, (long)kLQ * 2 * kH,
        (long)kLD * 3 * kH, 3 * kH, 0,
        0, 0, 0);

    // 7) last = doc_output[b, doc_lens[b]-1, :]
    gather_last<<<dim3(kB), blk, 0, stream>>>(doc_out, dlens, last_out);
}
